// DeepGcnV2_67448166416658
// MI455X (gfx1250) — matrix operand order repaired and checked by reading, not compiled, not run on hardware
//
#include <hip/hip_runtime.h>
#include <hip/hip_bf16.h>
#include <cmath>

#define BS 32
#define NN 2048
#define IN_SIZE 128
#define HID 64
#define NLAYERS 4
#define NW 64              // u32 words of adjacency bits per row (NN/32)

typedef __bf16 bf16_t;
typedef __attribute__((ext_vector_type(16))) __bf16 bf16x16;
typedef __attribute__((ext_vector_type(8)))  float  f32x8;

union AFrag { bf16x16 v; bf16_t e[16]; unsigned u[8]; uint4 q[2]; };
union CFrag { f32x8 v; float f[8]; };

__device__ inline f32x8 wmma_bf16f32(bf16x16 a, bf16x16 b, f32x8 c) {
  // D = A(16x32 bf16) x B(32x16 bf16) + C(16x16 f32)
  return __builtin_amdgcn_wmma_f32_16x16x32_bf16(false, a, false, b, (short)0, c, false, false);
}

__device__ inline f32x8 f8zero() {
  f32x8 z = {0.f, 0.f, 0.f, 0.f, 0.f, 0.f, 0.f, 0.f};
  return z;
}

__device__ inline float wave_sum(float v) {
  #pragma unroll
  for (int m = 16; m >= 1; m >>= 1) v += __shfl_xor(v, m, 32);
  return v;
}

// ---------------------------------------------------------------------------
// Pack a row-major f32 K x N matrix into WMMA bf16 B-fragment layout:
// dst[((kt*NG + ng)*32 + lane)*16 + j] <- src[(kt*32 + (lane/16)*16 + j)*N + ng*16 + lane%16]
// ---------------------------------------------------------------------------
__global__ void k_pack_w(const float* __restrict__ src, bf16_t* __restrict__ dst,
                         int K, int Nc) {
  int t = blockIdx.x * blockDim.x + threadIdx.x;
  if (t >= K * Nc) return;
  int NG = Nc >> 4;
  int j    = t & 15;
  int lane = (t >> 4) & 31;
  int ng   = (t >> 9) % NG;
  int kt   = t / (512 * NG);
  int kk = kt * 32 + (lane >> 4) * 16 + j;
  int n  = ng * 16 + (lane & 15);
  dst[t] = (bf16_t)src[kk * Nc + n];
}

// ---------------------------------------------------------------------------
// Fused: adjacency -> bitmask (1 bit/edge via __ballot) + dinv = rsqrt(deg+1).
// One wave32 per row; reads adj f32 exactly once for the whole call.
// Bit c%32 of word abits[row*64 + c/32] == (adj[row][c] != 0).
// ---------------------------------------------------------------------------
__global__ void k_adj_pack(const float* __restrict__ adj, unsigned* __restrict__ abits,
                           float* __restrict__ dinv) {
  int row  = blockIdx.x * 8 + (threadIdx.x >> 5);
  int lane = threadIdx.x & 31;
  const float* ar = adj + (size_t)row * NN;
  unsigned w0 = 0, w1 = 0;
  int cnt = 0;
  for (int ch = 0; ch < NW; ++ch) {
    float v = ar[ch * 32 + lane];
    unsigned mk = (unsigned)__ballot(v != 0.0f);
    cnt += __popc(mk);                 // identical in all lanes
    if (lane == (ch & 31)) { if (ch < 32) w0 = mk; else w1 = mk; }
  }
  unsigned* abRow = abits + (size_t)row * NW;
  abRow[lane]      = w0;
  abRow[32 + lane] = w1;
  if (lane == 0) dinv[row] = rsqrtf((float)cnt + 1.0f);
}

// ---------------------------------------------------------------------------
// h = x @ proj_w + proj_b  via bf16 WMMA, f32 accumulate.
// Block: 256 thr = 8 waves; 32-row M-block, full N=64. Waves = 2 mtiles x 4 ntiles.
// ---------------------------------------------------------------------------
__global__ void k_proj(const float* __restrict__ x, const bf16_t* __restrict__ pw,
                       const float* __restrict__ pb, float* __restrict__ h) {
  int lane = threadIdx.x & 31;
  int w    = threadIdx.x >> 5;
  int mt = w >> 2, nt = w & 3;
  int r0 = blockIdx.x * 32;
  int m  = r0 + 16 * mt + (lane & 15);
  const float* xr = x + (size_t)m * IN_SIZE;
  int lo = (lane >> 4) * 16;
  CFrag c; c.v = f8zero();
  for (int k0 = 0; k0 < IN_SIZE; k0 += 32) {
    union { float4 f4[4]; float f[16]; } t;
    t.f4[0] = *(const float4*)(xr + k0 + lo);
    t.f4[1] = *(const float4*)(xr + k0 + lo + 4);
    t.f4[2] = *(const float4*)(xr + k0 + lo + 8);
    t.f4[3] = *(const float4*)(xr + k0 + lo + 12);
    AFrag a;
    #pragma unroll
    for (int i = 0; i < 8; ++i) {
      a.e[i]     = (bf16_t)t.f[i];
      a.e[8 + i] = (bf16_t)t.f[8 + i];
    }
    AFrag b;
    const bf16_t* bp = pw + (((k0 >> 5) * 4 + nt) * 32 + lane) * 16;
    b.q[0] = *(const uint4*)bp;
    b.q[1] = *(const uint4*)(bp + 8);
    c.v = wmma_bf16f32(a.v, b.v, c.v);
  }
  #pragma unroll
  for (int r = 0; r < 8; ++r) {
    int mm = r0 + 16 * mt + r + 8 * (lane >> 4);
    int n  = 16 * nt + (lane & 15);
    h[(size_t)mm * HID + n] = c.f[r] + pb[n];
  }
}

// ---------------------------------------------------------------------------
// z = relu(layer_norm(h)); writes BOTH z' = dinv*z (B-fragment layout, for the
// binary-A WMMA) and plain z (same layout, for the 0.1*z term).
// One wave32 per node row (64 features -> 2 per lane).
// ---------------------------------------------------------------------------
__global__ void k_ln_relu(const float* __restrict__ h, const float* __restrict__ g,
                          const float* __restrict__ bb, const float* __restrict__ dinv,
                          bf16_t* __restrict__ zs, bf16_t* __restrict__ zpl) {
  int row  = blockIdx.x * 8 + (threadIdx.x >> 5);
  int lane = threadIdx.x & 31;
  const float* hr = h + (size_t)row * HID;
  float2 v = *(const float2*)(hr + 2 * lane);
  float mu = wave_sum(v.x + v.y) * (1.0f / 64.0f);
  float dx = v.x - mu, dy = v.y - mu;
  float var = wave_sum(dx * dx + dy * dy) * (1.0f / 64.0f);
  float rs = rsqrtf(var + 1e-5f);
  int n0 = 2 * lane, n1 = 2 * lane + 1;
  float z0 = fmaxf(dx * rs * g[n0] + bb[n0], 0.0f);
  float z1 = fmaxf(dy * rs * g[n1] + bb[n1], 0.0f);
  float di = dinv[row];
  int b = row >> 11, i = row & 2047;
  int kt = i >> 5, kl = i & 31;
  size_t base = (size_t)b * (NN * HID);
  int lp = (kl >> 4) * 16;
  int j  = kl & 15;
  size_t i0 = base + ((size_t)((kt * 4 + (n0 >> 4)) * 32 + lp + (n0 & 15))) * 16 + j;
  size_t i1 = base + ((size_t)((kt * 4 + (n1 >> 4)) * 32 + lp + (n1 & 15))) * 16 + j;
  zs[i0]  = (bf16_t)(z0 * di);
  zs[i1]  = (bf16_t)(z1 * di);
  zpl[i0] = (bf16_t)z0;
  zpl[i1] = (bf16_t)z1;
}

// ---------------------------------------------------------------------------
// Fused GCN2 layer with binary adjacency:
//   acc = A @ z'            (A expanded from bitmask to bf16 {0,1} in registers)
//   p   = dinv_m * (acc + z'_m)          == adj_hat @ z
//   s   = 0.9*p + 0.1*z
//   h  += (1-beta)*s + beta*(s @ W)
// Block: 256 thr = 8 waves, owns 32 output rows x 64 cols of one batch.
// ---------------------------------------------------------------------------
__global__ void k_layer(const unsigned* __restrict__ abits, const float* __restrict__ dinv,
                        const bf16_t* __restrict__ zs, const bf16_t* __restrict__ zpl,
                        const bf16_t* __restrict__ cw, float* __restrict__ h, float beta) {
  __shared__ float ss[32][68];      // s tile, padded vs bank conflicts

  int b  = blockIdx.x >> 6;
  int m0 = (blockIdx.x & 63) * 32;
  int lane = threadIdx.x & 31;
  int w    = threadIdx.x >> 5;
  int mt = w >> 2, nt = w & 3;
  int lo = (lane >> 4) * 8;
  int m  = m0 + 16 * mt + (lane & 15);

  const unsigned* arow = abits + ((size_t)b * NN + m) * NW;
  const bf16_t*   zsb  = zs + (size_t)b * (NN * HID);

  CFrag c; c.v = f8zero();
  for (int k0 = 0; k0 < NN; k0 += 128) {           // 4 K-steps per 128-bit load
    uint4 wq = *(const uint4*)(arow + (k0 >> 5));
    unsigned wv[4] = {wq.x, wq.y, wq.z, wq.w};
    #pragma unroll
    for (int s4 = 0; s4 < 4; ++s4) {
      unsigned w0 = wv[s4] >> (2 * lo);            // 16 contiguous k bits at 0..15
      AFrag a;
      #pragma unroll
      for (int p = 0; p < 4; ++p) {                // build packed bf16 {0,1} pairs
        a.u[p]     = (((w0 >> (2 * p)) & 1u)      ? 0x00003F80u : 0u) |
                     (((w0 >> (2 * p + 1)) & 1u)  ? 0x3F800000u : 0u);
        a.u[4 + p] = (((w0 >> (8 + 2 * p)) & 1u) ? 0x00003F80u : 0u) |
                     (((w0 >> (9 + 2 * p)) & 1u) ? 0x3F800000u : 0u);
      }
      AFrag bf;
      const bf16_t* bp = zsb + ((((k0 >> 5) + s4) * 4 + nt) * 32 + lane) * 16;
      bf.q[0] = *(const uint4*)bp;
      bf.q[1] = *(const uint4*)(bp + 8);
      c.v = wmma_bf16f32(a.v, bf.v, c.v);
    }
  }

  // epilogue: p = dinv_m*(acc + z'_m);  s = 0.9*p + 0.1*z
  const float* dvp = dinv + b * NN + m0 + 16 * mt + 8 * (lane >> 4);
  float4 dv0 = *(const float4*)dvp;
  float4 dv1 = *(const float4*)(dvp + 4);
  float dmv[8] = {dv0.x, dv0.y, dv0.z, dv0.w, dv1.x, dv1.y, dv1.z, dv1.w};
  union { uint4 q; bf16_t e[8]; } zo, zl;
  size_t goff = ((size_t)(((m0 >> 5) * 4 + nt) * 32 + 16 * mt + (lane & 15))) * 16 + lo;
  zo.q = *(const uint4*)(zsb + goff);
  zl.q = *(const uint4*)(zpl + (size_t)b * (NN * HID) + goff);
  #pragma unroll
  for (int r = 0; r < 8; ++r) {
    float p  = dmv[r] * (c.f[r] + (float)zo.e[r]);
    float sv = 0.9f * p + 0.1f * (float)zl.e[r];
    ss[16 * mt + r + 8 * (lane >> 4)][16 * nt + (lane & 15)] = sv;
  }
  __syncthreads();

  // SW = s @ W  (K = 64 -> two WMMAs), W pre-packed per layer
  CFrag d2; d2.v = f8zero();
  #pragma unroll
  for (int kt = 0; kt < 2; ++kt) {
    AFrag a2;
    int ra  = 16 * mt + (lane & 15);
    int cb2 = kt * 32 + 2 * lo;
    #pragma unroll
    for (int i = 0; i < 8; ++i) {
      a2.e[i]     = (bf16_t)ss[ra][cb2 + i];
      a2.e[8 + i] = (bf16_t)ss[ra][cb2 + 8 + i];
    }
    AFrag b2;
    const bf16_t* bp2 = cw + ((kt * 4 + nt) * 32 + lane) * 16;
    b2.q[0] = *(const uint4*)bp2;
    b2.q[1] = *(const uint4*)(bp2 + 8);
    d2.v = wmma_bf16f32(a2.v, b2.v, d2.v);
  }

  // h += (1-beta)*s + beta*SW
  #pragma unroll
  for (int r = 0; r < 8; ++r) {
    int lr = 16 * mt + r + 8 * (lane >> 4);
    int n  = 16 * nt + (lane & 15);
    size_t idx = ((size_t)b * NN + m0 + lr) * HID + n;
    h[idx] = h[idx] + (1.0f - beta) * ss[lr][n] + beta * d2.f[r];
  }
}

// ---------------------------------------------------------------------------
// out = h @ head_w + head_b   (N=1 -> wave dot product per row)
// ---------------------------------------------------------------------------
__global__ void k_head(const float* __restrict__ h, const float* __restrict__ hw,
                       const float* __restrict__ hb, float* __restrict__ out) {
  int row  = blockIdx.x * 8 + (threadIdx.x >> 5);
  int lane = threadIdx.x & 31;
  const float* hr = h + (size_t)row * HID;
  float2 v  = *(const float2*)(hr + 2 * lane);
  float2 wv = *(const float2*)(hw + 2 * lane);
  float s = wave_sum(v.x * wv.x + v.y * wv.y);
  if (lane == 0) out[row] = s + hb[0];
}

extern "C" void kernel_launch(void* const* d_in, const int* in_sizes, int n_in,
                              void* d_out, int out_size, void* d_ws, size_t ws_size,
                              hipStream_t stream) {
  (void)in_sizes; (void)n_in; (void)out_size; (void)ws_size;
  const float* x      = (const float*)d_in[0];
  const float* adj    = (const float*)d_in[1];
  const float* proj_w = (const float*)d_in[2];
  const float* proj_b = (const float*)d_in[3];
  const float* ln_g   = (const float*)d_in[4];
  const float* ln_b   = (const float*)d_in[5];
  const float* conv_w = (const float*)d_in[6];
  const float* head_w = (const float*)d_in[7];
  const float* head_b = (const float*)d_in[8];
  float* out = (float*)d_out;

  // workspace layout (~50.6 MB)
  char* ws = (char*)d_ws;
  const size_t sz_h  = (size_t)BS * NN * HID * 4;     // 16.78 MB f32 hidden state
  const size_t sz_z  = (size_t)BS * NN * HID * 2;     //  8.39 MB packed bf16 (x2)
  const size_t sz_di = (size_t)BS * NN * 4;           //  0.26 MB dinv
  const size_t sz_ab = (size_t)BS * NN * NW * 4;      // 16.78 MB adjacency bitmask
  float*    h    = (float*)ws;
  bf16_t*   zsc  = (bf16_t*)(ws + sz_h);
  bf16_t*   zpl  = (bf16_t*)(ws + sz_h + sz_z);
  float*    dinv = (float*)(ws + sz_h + 2 * sz_z);
  unsigned* ab   = (unsigned*)(ws + sz_h + 2 * sz_z + sz_di);
  bf16_t*   pw   = (bf16_t*)(ws + sz_h + 2 * sz_z + sz_di + sz_ab);
  bf16_t*   cw   = pw + (size_t)IN_SIZE * HID;

  k_pack_w<<<(IN_SIZE * HID) / 256, 256, 0, stream>>>(proj_w, pw, IN_SIZE, HID);
  for (int l = 0; l < NLAYERS; ++l)
    k_pack_w<<<(HID * HID) / 256, 256, 0, stream>>>(conv_w + l * HID * HID,
                                                    cw + l * HID * HID, HID, HID);
  k_adj_pack<<<(BS * NN) / 8, 256, 0, stream>>>(adj, ab, dinv);
  k_proj<<<(BS * NN) / 32, 256, 0, stream>>>(x, pw, proj_b, h);
  for (int l = 0; l < NLAYERS; ++l) {
    float beta = logf(1.0f / (float)(l + 1) + 1.0f);
    k_ln_relu<<<(BS * NN) / 8, 256, 0, stream>>>(h, ln_g + l * HID, ln_b + l * HID,
                                                 dinv, zsc, zpl);
    k_layer<<<BS * (NN / 32), 256, 0, stream>>>(ab, dinv, zsc, zpl,
                                                cw + l * HID * HID, h, beta);
  }
  k_head<<<(BS * NN) / 8, 256, 0, stream>>>(h, head_w, head_b, out);
}
